// FirDecoder_11570641896041
// MI455X (gfx1250) — compile-verified, matrix operand order checked by reading, not run on hardware
//
#include <hip/hip_runtime.h>
#include <hip/hip_bf16.h>
#include <math.h>

// ---------------- types for WMMA ----------------
typedef __bf16 bf16_t;
typedef bf16_t v16bf __attribute__((ext_vector_type(16)));
typedef bf16_t v8bf  __attribute__((ext_vector_type(8)));
typedef float  v8f   __attribute__((ext_vector_type(8)));

__device__ __forceinline__ bf16_t f2bf(float f) {
    unsigned u = __builtin_bit_cast(unsigned, f);
    unsigned r = u + 0x7FFFu + ((u >> 16) & 1u);   // round-to-nearest-even
    unsigned short h = (unsigned short)(r >> 16);
    return __builtin_bit_cast(bf16_t, h);
}

// ---------------- constants ----------------
#define SRr   48000
#define FRAME 960
#define NFFT  3840
#define NWIN  1024
#define NFREQ 1921
#define BB    8
#define LFf   200
#define CINT  512
#define CINP  480               // padded input-feature count (450 -> 480)
#define NPOS  (BB*LFf)          // 1600
#define LW    (LFf*FRAME)       // 192000
#define LPAD  (NFFT + LW)       // 195840
#define PI_F  3.14159265358979f

// =====================================================================
// WMMA GEMM, bf16 x bf16 -> f32:  Y(N,ldy) = act( A(MxK) * B + bias )[+res]
//   A: M x K row-major bf16 (lda, multiple of 8; K multiple of 32)
//   B: N x K row-major bf16 (activations^T)
//   Each wave: one 16(M) x 64(N) strip -> 1 A-frag reused by 4 B-frags.
//   A frag (16x32 bf16): lane m=lane&15, hi=lane>>4; elements 0..7 = K run
//   [kk+hi*8,+8), elements 8..15 = [kk+16+hi*8,+8)  -> two b128 loads.
//   D tile: lane n = lane&15, VGPR r -> m = r + 8*hi.
// =====================================================================
__global__ void wmma_gemm_kernel(const bf16_t* __restrict__ A, int lda,
                                 const bf16_t* __restrict__ Bm, int ldb,
                                 float* __restrict__ Yf, bf16_t* __restrict__ Yb,
                                 int ldy,
                                 const float* __restrict__ bias,
                                 const float* __restrict__ res,
                                 int M, int N, int K, int act)
{
    int wave = threadIdx.x >> 5;
    int lane = threadIdx.x & 31;
    int mtile = blockIdx.y * 4 + wave;
    if (mtile * 16 >= M) return;            // wave-uniform: EXEC stays all-1s

    int rowi = lane & 15;
    int hi   = lane >> 4;
    int mA   = mtile * 16 + rowi;
    if (mA >= M) mA = M - 1;                // clamp: loads valid, stores skipped
    const bf16_t* Arow = A + (size_t)mA * lda + hi * 8;

    int n0 = blockIdx.x * 64;
    const bf16_t* Br0 = Bm + (size_t)(n0 + rowi) * ldb + hi * 8;
    const bf16_t* Br1 = Br0 + (size_t)16 * ldb;
    const bf16_t* Br2 = Br1 + (size_t)16 * ldb;
    const bf16_t* Br3 = Br2 + (size_t)16 * ldb;

    v8f acc0 = {}, acc1 = {}, acc2 = {}, acc3 = {};
    for (int kk = 0; kk < K; kk += 32) {
        v8bf alo = *(const v8bf*)(Arow + kk);
        v8bf ahi = *(const v8bf*)(Arow + kk + 16);
        v16bf af = __builtin_shufflevector(alo, ahi,
                     0,1,2,3,4,5,6,7,8,9,10,11,12,13,14,15);

        v8bf b0l = *(const v8bf*)(Br0 + kk);
        v8bf b0h = *(const v8bf*)(Br0 + kk + 16);
        v8bf b1l = *(const v8bf*)(Br1 + kk);
        v8bf b1h = *(const v8bf*)(Br1 + kk + 16);
        v8bf b2l = *(const v8bf*)(Br2 + kk);
        v8bf b2h = *(const v8bf*)(Br2 + kk + 16);
        v8bf b3l = *(const v8bf*)(Br3 + kk);
        v8bf b3h = *(const v8bf*)(Br3 + kk + 16);

        v16bf bf0 = __builtin_shufflevector(b0l, b0h, 0,1,2,3,4,5,6,7,8,9,10,11,12,13,14,15);
        v16bf bf1 = __builtin_shufflevector(b1l, b1h, 0,1,2,3,4,5,6,7,8,9,10,11,12,13,14,15);
        v16bf bf2 = __builtin_shufflevector(b2l, b2h, 0,1,2,3,4,5,6,7,8,9,10,11,12,13,14,15);
        v16bf bf3 = __builtin_shufflevector(b3l, b3h, 0,1,2,3,4,5,6,7,8,9,10,11,12,13,14,15);

        acc0 = __builtin_amdgcn_wmma_f32_16x16x32_bf16(false, af, false, bf0, (short)0, acc0, false, false);
        acc1 = __builtin_amdgcn_wmma_f32_16x16x32_bf16(false, af, false, bf1, (short)0, acc1, false, false);
        acc2 = __builtin_amdgcn_wmma_f32_16x16x32_bf16(false, af, false, bf2, (short)0, acc2, false, false);
        acc3 = __builtin_amdgcn_wmma_f32_16x16x32_bf16(false, af, false, bf3, (short)0, acc3, false, false);
    }

    v8f accs[4] = {acc0, acc1, acc2, acc3};
    #pragma unroll
    for (int t = 0; t < 4; ++t) {
        int n = n0 + t * 16 + rowi;
        #pragma unroll
        for (int r = 0; r < 8; ++r) {
            int m = mtile * 16 + r + (hi << 3);
            if (m >= M) continue;
            float v = accs[t][r] + (bias ? bias[m] : 0.f);
            if (act == 1)      v = 0.5f * v * (1.f + erff(v * 0.70710678f)); // exact gelu
            else if (act == 2) v = (v > 0.f) ? (v + 1.f) : __expf(v);        // elu+1
            if (res) v += res[(size_t)n * ldy + m];
            if (Yb) Yb[(size_t)n * ldy + m] = f2bf(v);
            else    Yf[(size_t)n * ldy + m] = v;
        }
    }
}

// =====================================================================
// Packing / conversion kernels (run once per launch; L2-trivial traffic)
// =====================================================================
__global__ void cvt_bf16(const float* __restrict__ src, bf16_t* __restrict__ dst, int count)
{
    int id = blockIdx.x * 256 + threadIdx.x;
    if (id < count) dst[id] = f2bf(src[id]);
}

__global__ void pack_inw(const float* __restrict__ w, bf16_t* __restrict__ dst)
{
    int id = blockIdx.x * 256 + threadIdx.x;
    if (id >= CINT * CINP) return;
    int m = id / CINP, k = id % CINP;
    dst[id] = f2bf(k < 450 ? w[m * 450 + k] : 0.f);
}

__global__ void pack_w2b(const float* __restrict__ w, bf16_t* __restrict__ w2)
{
    int id = blockIdx.x * 256 + threadIdx.x;
    if (id >= NWIN * 3584) return;
    int o = id / 3584, k = id % 3584;
    int t = k >> 9, c = k & 511;
    w2[id] = f2bf(w[((size_t)(o * 512 + c)) * 7 + t]);
}

__global__ void build_xin(const float* __restrict__ content, const float* __restrict__ f0,
                          const float* __restrict__ energy,  const float* __restrict__ spk,
                          bf16_t* __restrict__ xin)
{
    int id = blockIdx.x * 256 + threadIdx.x;
    if (id >= NPOS * CINP) return;
    int n = id / CINP, c = id % CINP;
    int b = n / LFf, l = n % LFf;
    float v;
    if (c < 192)        v = content[(b * 192 + c) * LFf + l];
    else if (c < 448)   v = spk[b * 256 + (c - 192)];
    else if (c == 448) { float fv = f0[b * LFf + l]; fv = fv > 0.f ? fv : 0.f; v = logf(fv + 1e-6f); }
    else if (c == 449)  v = energy[b * LFf + l];
    else                v = 0.f;                       // K pad 450 -> 480
    xin[(size_t)n * CINP + c] = f2bf(v);
}

// =====================================================================
// Trunk elementwise kernels
// =====================================================================
__global__ void dwconv_k(const float* __restrict__ x, const float* __restrict__ w,
                         const float* __restrict__ bias, float* __restrict__ h)
{
    int id = blockIdx.x * 256 + threadIdx.x;
    if (id >= NPOS * CINT) return;
    int n = id >> 9, c = id & 511;
    int b = n / LFf, l = n % LFf;
    float acc = bias[c];
    #pragma unroll
    for (int t = 0; t < 7; ++t) {
        int ll = l + t - 3;
        if (ll >= 0 && ll < LFf)
            acc += x[((size_t)(b * LFf + ll) << 9) + c] * w[c * 7 + t];
    }
    h[id] = acc;
}

// channel LayerNorm; optional f32 and/or bf16 outputs
__global__ void chan_ln(const float* __restrict__ x, float* __restrict__ yf,
                        bf16_t* __restrict__ yb,
                        const float* __restrict__ g, const float* __restrict__ bta)
{
    __shared__ float s1[256], s2[256];
    int n = blockIdx.x;
    const float* row = x + (size_t)n * CINT;
    float v0 = row[threadIdx.x], v1 = row[threadIdx.x + 256];
    s1[threadIdx.x] = v0 + v1;
    s2[threadIdx.x] = v0 * v0 + v1 * v1;
    __syncthreads();
    for (int s = 128; s > 0; s >>= 1) {
        if (threadIdx.x < s) { s1[threadIdx.x] += s1[threadIdx.x + s]; s2[threadIdx.x] += s2[threadIdx.x + s]; }
        __syncthreads();
    }
    float mean = s1[0] * (1.f / CINT);
    float var  = s2[0] * (1.f / CINT) - mean * mean;
    float rs   = rsqrtf(var + 1e-5f);
    int c0 = threadIdx.x, c1 = threadIdx.x + 256;
    float o0 = (v0 - mean) * rs * g[c0] + bta[c0];
    float o1 = (v1 - mean) * rs * g[c1] + bta[c1];
    if (yf) { yf[(size_t)n * CINT + c0] = o0; yf[(size_t)n * CINT + c1] = o1; }
    if (yb) { yb[(size_t)n * CINT + c0] = f2bf(o0); yb[(size_t)n * CINT + c1] = f2bf(o1); }
}

// ------------- harmonic oscillator -------------
__device__ __forceinline__ float fs_at(const float* f0, int b, int t)
{
    float pos = (t + 0.5f) * (1.f / FRAME) - 0.5f;
    pos = fminf(fmaxf(pos, 0.f), (float)(LFf - 1));
    int i0 = (int)floorf(pos);
    int i1 = i0 + 1 > LFf - 1 ? LFf - 1 : i0 + 1;
    float w = pos - (float)i0;
    return f0[b * LFf + i0] * (1.f - w) + f0[b * LFf + i1] * w;
}

__global__ void fs_block_sums(const float* __restrict__ f0, float* __restrict__ part)
{
    __shared__ float s1[256];
    int blk = blockIdx.x, b = blk / LFf, B0 = blk % LFf;
    float acc = 0.f;
    for (int j = threadIdx.x; j < FRAME; j += 256) acc += fs_at(f0, b, B0 * FRAME + j);
    s1[threadIdx.x] = acc; __syncthreads();
    for (int s = 128; s > 0; s >>= 1) {
        if (threadIdx.x < s) s1[threadIdx.x] += s1[threadIdx.x + s];
        __syncthreads();
    }
    if (threadIdx.x == 0) part[blk] = s1[0];
}

__global__ void scan_partials(const float* __restrict__ part, float* __restrict__ phs)
{
    if (threadIdx.x != 0) return;
    int b = blockIdx.x;
    float run = 0.f;
    for (int i = 0; i < LFf; ++i) { phs[b * LFf + i] = run; run += part[b * LFf + i]; }
}

__global__ void harm_fill(const float* __restrict__ f0, const float* __restrict__ amps,
                          const float* __restrict__ phs, float* __restrict__ dsp)
{
    __shared__ float fsb[FRAME];
    int blk = blockIdx.x, b = blk / LFf, B0 = blk % LFf;
    for (int j = threadIdx.x; j < FRAME; j += 256) fsb[j] = fs_at(f0, b, B0 * FRAME + j);
    __syncthreads();
    if (threadIdx.x == 0) {
        float run = phs[blk];
        for (int j = 0; j < FRAME; ++j) { run += fsb[j]; fsb[j] = run; }
    }
    __syncthreads();
    float uv = (f0[b * LFf + B0] > 20.f) ? 1.f : 0.f;
    for (int j = threadIdx.x; j < FRAME; j += 256) {
        int t = B0 * FRAME + j;
        float fsv   = fs_at(f0, b, t);
        float phase = fsb[j] * (PI_F / (float)SRr);
        float pix   = fmodf(phase, PI_F);
        float av    = roundf((float)SRr / fmaxf(fsv, 20.f) * 0.5f) * 2.f + 1.f;
        float h     = (pix < 1e-8f) ? 1.f : (sinf(av * pix) / (av * sinf(pix)));
        float pos = (t + 0.5f) * (1.f / FRAME) - 0.5f;
        pos = fminf(fmaxf(pos, 0.f), (float)(LFf - 1));
        int i0 = (int)floorf(pos);
        int i1 = i0 + 1 > LFf - 1 ? LFf - 1 : i0 + 1;
        float w = pos - (float)i0;
        float ampv = amps[b * LFf + i0] * (1.f - w) + amps[b * LFf + i1] * w;
        dsp[b * LW + t] = h * uv * ampv;
    }
}

// ------------- noise + STFT shaping -------------
__device__ __forceinline__ float rng01(unsigned idx)
{
    unsigned s = idx * 747796405u + 2891336453u + 123u;
    unsigned w = ((s >> ((s >> 28) + 4u)) ^ s) * 277803737u;
    w = (w >> 22) ^ w;
    return (float)(w >> 8) * (1.f / 16777216.f);
}

__global__ void gen_xp(float* __restrict__ xp)
{
    int id = blockIdx.x * 256 + threadIdx.x;
    if (id >= BB * LPAD) return;
    int b = id / LPAD, p = id % LPAD;
    int q = p - NFFT / 2;
    if (q < 0) q = -q;
    else if (q >= LW) q = 2 * LW - 2 - q;     // reflect pad
    xp[id] = rng01((unsigned)(b * LW + q));
}

__global__ void twiddle_init(float* __restrict__ tw)
{
    int t = blockIdx.x * 256 + threadIdx.x;
    if (t >= NFFT) return;
    float ang = 6.283185307179586f * (float)t / (float)NFFT;
    tw[t] = cosf(ang);
    tw[NFFT + t] = sinf(ang);
}

// forward DFT per frame, fused with spectral-kernel multiply (frame 0 -> 0)
__global__ void stft_kernel(const float* __restrict__ xp, const float* __restrict__ tw,
                            const float* __restrict__ ker, float* __restrict__ spec)
{
    __shared__ float tc[NFFT], ts[NFFT], xw[NFFT];
    int i = blockIdx.x, b = blockIdx.y;
    for (int t = threadIdx.x; t < NFFT; t += 256) {
        float c = tw[t], s = tw[NFFT + t];
        tc[t] = c; ts[t] = s;
        xw[t] = xp[(size_t)b * LPAD + i * FRAME + t] * (0.5f - 0.5f * c);
    }
    __syncthreads();
    for (int k = threadIdx.x; k < NFREQ; k += 256) {
        float re = 0.f, im = 0.f;
        int idx = 0;
        for (int t = 0; t < NFFT; ++t) {
            float xv = xw[t];
            re += xv * tc[idx];
            im -= xv * ts[idx];
            idx += k; if (idx >= NFFT) idx -= NFFT;
        }
        float f = (i == 0) ? 0.f : ker[(size_t)(b * LFf + (i - 1)) * NFREQ + k];
        size_t o = (((size_t)b * (LFf + 1) + i) * NFREQ + k) * 2;
        spec[o] = re * f; spec[o + 1] = im * f;
    }
}

// inverse real DFT per frame, times window
__global__ void istft_kernel(const float* __restrict__ spec, const float* __restrict__ tw,
                             float* __restrict__ ft)
{
    __shared__ float tc[NFFT], ts[NFFT];
    __shared__ float sre[NFREQ], sim[NFREQ];
    int i = blockIdx.x, b = blockIdx.y;
    for (int t = threadIdx.x; t < NFFT; t += 256) { tc[t] = tw[t]; ts[t] = tw[NFFT + t]; }
    for (int k = threadIdx.x; k < NFREQ; k += 256) {
        size_t o = (((size_t)b * (LFf + 1) + i) * NFREQ + k) * 2;
        sre[k] = spec[o]; sim[k] = spec[o + 1];
    }
    __syncthreads();
    const float inv = 1.f / (float)NFFT;
    for (int t = threadIdx.x; t < NFFT; t += 256) {
        float acc = sre[0] + ((t & 1) ? -sre[NFREQ - 1] : sre[NFREQ - 1]);
        int id = 0;
        for (int k = 1; k < NFREQ - 1; ++k) {
            id += t; if (id >= NFFT) id -= NFFT;
            acc += 2.f * (sre[k] * tc[id] - sim[k] * ts[id]);
        }
        float wv = 0.5f - 0.5f * tc[t];
        ft[((size_t)b * (LFf + 1) + i) * NFFT + t] = acc * inv * wv;
    }
}

// overlap-add with analytic window normalization; adds into dsp (harm already there)
__global__ void ola_add(const float* __restrict__ ft, const float* __restrict__ tw,
                        float* __restrict__ dsp)
{
    int id = blockIdx.x * 256 + threadIdx.x;
    if (id >= BB * LPAD) return;
    int b = id / LPAD, p = id % LPAD;
    int ihi = p / FRAME; if (ihi > LFf) ihi = LFf;
    int ilo = (p >= NFFT) ? ((p - (NFFT - 1) + FRAME - 1) / FRAME) : 0;
    float sum = 0.f, wn = 0.f;
    for (int i = ilo; i <= ihi; ++i) {
        int off = p - i * FRAME;
        float wv = 0.5f - 0.5f * tw[off];
        sum += ft[((size_t)b * (LFf + 1) + i) * NFFT + off];
        wn  += wv * wv;
    }
    float y = sum / ((wn > 1e-11f) ? wn : 1.f);
    int q = p - NFFT / 2;
    if (q >= 0 && q < LW) dsp[(size_t)b * LW + q] += y;
}

// ------------- filter head helpers -------------
__global__ void im2col_edge(const bf16_t* __restrict__ x, bf16_t* __restrict__ xe)
{
    int id = blockIdx.x * 256 + threadIdx.x;
    if (id >= NPOS * 3584) return;
    int n = id / 3584, k = id % 3584;
    int t = k >> 9, c = k & 511;
    int b = n / LFf, l = n % LFf;
    int ll = l + t - 3;
    ll = ll < 0 ? 0 : (ll > LFf - 1 ? LFf - 1 : ll);
    xe[id] = x[((size_t)(b * LFf + ll) << 9) + c];
}

// per-frame FIR + fold, gather style (deterministic, fully overwrites out)
__global__ void fir_fold(const float* __restrict__ dsp, const float* __restrict__ filt,
                         float* __restrict__ out)
{
    __shared__ float fl[3][NWIN];
    __shared__ float df[3][FRAME];
    int b = blockIdx.x / LFf, L0 = blockIdx.x % LFf;
    for (int d = 0; d < 3; ++d) {
        int l = L0 - 2 + d;
        for (int k = threadIdx.x; k < NWIN; k += 256)
            fl[d][k] = (l >= 0) ? filt[((size_t)(b * LFf + l) * NWIN) + k] : 0.f;
        for (int s = threadIdx.x; s < FRAME; s += 256)
            df[d][s] = (l >= 0) ? dsp[(size_t)b * LW + l * FRAME + s] : 0.f;
    }
    __syncthreads();
    for (int j0 = threadIdx.x; j0 < FRAME; j0 += 256) {
        float acc = 0.f;
        #pragma unroll
        for (int d = 0; d < 3; ++d) {
            int j = j0 + (2 - d) * FRAME;
            if (j >= FRAME + NWIN) continue;
            int s0 = j - NWIN; if (s0 < 0) s0 = 0;
            int s1 = j - 1;    if (s1 > FRAME - 1) s1 = FRAME - 1;
            for (int s = s0; s <= s1; ++s)
                acc += df[d][s] * fl[d][NWIN + s - j];
        }
        out[(size_t)b * LW + L0 * FRAME + j0] = acc;
    }
}

// =====================================================================
// Host orchestration
// =====================================================================
extern "C" void kernel_launch(void* const* d_in, const int* in_sizes, int n_in,
                              void* d_out, int out_size, void* d_ws, size_t ws_size,
                              hipStream_t stream)
{
    (void)in_sizes; (void)n_in; (void)out_size; (void)ws_size;

    const float* content    = (const float*)d_in[0];
    const float* f0         = (const float*)d_in[1];
    const float* energy     = (const float*)d_in[2];
    const float* spk        = (const float*)d_in[3];
    const float* s_in_w     = (const float*)d_in[4];
    const float* s_in_b     = (const float*)d_in[5];
    const float* s_dw_w     = (const float*)d_in[6];
    const float* s_dw_b     = (const float*)d_in[7];
    const float* s_ln_p     = (const float*)d_in[8];
    const float* s_pw1_w    = (const float*)d_in[9];
    const float* s_pw1_b    = (const float*)d_in[10];
    const float* s_pw2_w    = (const float*)d_in[11];
    const float* s_pw2_b    = (const float*)d_in[12];
    const float* s_out_norm = (const float*)d_in[13];
    const float* f_in_w     = (const float*)d_in[14];
    const float* f_in_b     = (const float*)d_in[15];
    const float* f_dw_w     = (const float*)d_in[16];
    const float* f_dw_b     = (const float*)d_in[17];
    const float* f_ln_p     = (const float*)d_in[18];
    const float* f_pw1_w    = (const float*)d_in[19];
    const float* f_pw1_b    = (const float*)d_in[20];
    const float* f_pw2_w    = (const float*)d_in[21];
    const float* f_pw2_b    = (const float*)d_in[22];
    const float* f_out_norm = (const float*)d_in[23];
    const float* s_in_norm  = (const float*)d_in[24];
    const float* s_amp_w    = (const float*)d_in[25];
    const float* s_amp_b    = (const float*)d_in[26];
    const float* s_ker_w    = (const float*)d_in[27];
    const float* s_ker_b    = (const float*)d_in[28];
    const float* f_out_w    = (const float*)d_in[29];
    const float* f_out_b    = (const float*)d_in[30];

    // ---------------- f32 workspace ----------------
    float* W    = (float*)d_ws;
    float* XS   = W;                              // 1600*512
    float* H    = XS   + (size_t)NPOS * CINT;     // 1600*512
    float* AMPS = H    + (size_t)NPOS * CINT;     // 1600
    float* KER  = AMPS + NPOS;                    // 1600*1921
    float* PART = KER  + (size_t)NPOS * NFREQ;    // 1600
    float* PHS  = PART + NPOS;                    // 1600
    float* DSP  = PHS  + NPOS;                    // 8*192000
    float* XP   = DSP  + (size_t)BB * LW;         // 8*195840
    float* TW   = XP   + (size_t)BB * LPAD;       // 2*3840
    float* SPEC = TW   + 2 * NFFT;                // 8*201*1921*2  (24.7 MB)
    float* FT   = SPEC + (size_t)BB * (LFf+1) * NFREQ * 2; // 8*201*3840
    float* F32_END = FT + (size_t)BB * (LFf+1) * NFFT;
    // aliases over spent f32 buffers (stream-ordered, safe):
    bf16_t* XEb = (bf16_t*)SPEC;                  // 1600*3584 bf16 (<= SPEC)
    float*  FILT= FT;                             // 1600*1024 f32 (<= FT)

    // ---------------- bf16 workspace ----------------
    bf16_t* BWS   = (bf16_t*)F32_END;
    bf16_t* XINb  = BWS;                          // 1600*480
    bf16_t* HB    = XINb  + (size_t)NPOS * CINP;  // 1600*512
    bf16_t* H2b   = HB    + (size_t)NPOS * CINT;  // 1600*1536
    bf16_t* sInWb = H2b   + (size_t)NPOS * 1536;  // 512*480
    bf16_t* fInWb = sInWb + (size_t)CINT * CINP;  // 512*480
    bf16_t* sPw1b = fInWb + (size_t)CINT * CINP;  // 6*1536*512
    bf16_t* sPw2b = sPw1b + (size_t)6 * 1536 * CINT;
    bf16_t* fPw1b = sPw2b + (size_t)6 * CINT * 1536;
    bf16_t* fPw2b = fPw1b + (size_t)6 * 1536 * CINT;
    bf16_t* ampB  = fPw2b + (size_t)6 * CINT * 1536; // 512
    bf16_t* kerB  = ampB  + CINT;                 // 1921*512
    bf16_t* W2b   = kerB  + (size_t)NFREQ * CINT; // 1024*3584

    auto gemm = [&](const bf16_t* A, int lda, const bf16_t* Bm, int ldb,
                    float* Yf, bf16_t* Yb, int ldy,
                    const float* bias, const float* res,
                    int M, int N, int K, int act) {
        dim3 g(N / 64, (M + 63) / 64);
        wmma_gemm_kernel<<<g, dim3(128), 0, stream>>>(A, lda, Bm, ldb, Yf, Yb, ldy,
                                                      bias, res, M, N, K, act);
    };
    auto ln = [&](const float* x, float* yf, bf16_t* yb, const float* gb) {
        chan_ln<<<NPOS, 256, 0, stream>>>(x, yf, yb, gb, gb + CINT);
    };
    auto cvt = [&](const float* s, bf16_t* dst, size_t cnt) {
        cvt_bf16<<<(int)((cnt + 255) / 256), 256, 0, stream>>>(s, dst, (int)cnt);
    };

    // ---- weight conversion (once per call; ~6us of HBM traffic) ----
    pack_inw<<<(CINT * CINP + 255) / 256, 256, 0, stream>>>(s_in_w, sInWb);
    pack_inw<<<(CINT * CINP + 255) / 256, 256, 0, stream>>>(f_in_w, fInWb);
    cvt(s_pw1_w, sPw1b, (size_t)6 * 1536 * CINT);
    cvt(s_pw2_w, sPw2b, (size_t)6 * CINT * 1536);
    cvt(f_pw1_w, fPw1b, (size_t)6 * 1536 * CINT);
    cvt(f_pw2_w, fPw2b, (size_t)6 * CINT * 1536);
    cvt(s_amp_w, ampB, CINT);
    cvt(s_ker_w, kerB, (size_t)NFREQ * CINT);
    pack_w2b<<<(NWIN * 3584 + 255) / 256, 256, 0, stream>>>(f_out_w, W2b);

    // ---- fused input features (bf16, K padded to 480) ----
    build_xin<<<(NPOS * CINP + 255) / 256, 256, 0, stream>>>(content, f0, energy, spk, XINb);

    // ---- source net trunk ----
    gemm(sInWb, CINP, XINb, CINP, XS, nullptr, CINT, s_in_b, nullptr, CINT, NPOS, CINP, 0);
    ln(XS, XS, nullptr, s_in_norm);
    for (int i = 0; i < 6; ++i) {
        dwconv_k<<<(NPOS * CINT + 255) / 256, 256, 0, stream>>>(
            XS, s_dw_w + (size_t)i * CINT * 7, s_dw_b + (size_t)i * CINT, H);
        ln(H, nullptr, HB, s_ln_p + (size_t)i * 2 * CINT);
        gemm(sPw1b + (size_t)i * 1536 * CINT, CINT, HB, CINT, nullptr, H2b, 1536,
             s_pw1_b + (size_t)i * 1536, nullptr, 1536, NPOS, CINT, 1);
        gemm(sPw2b + (size_t)i * CINT * 1536, 1536, H2b, 1536, XS, nullptr, CINT,
             s_pw2_b + (size_t)i * CINT, XS, CINT, NPOS, 1536, 0);
    }
    ln(XS, nullptr, HB, s_out_norm);
    gemm(ampB, CINT, HB, CINT, AMPS, nullptr, 1, s_amp_b, nullptr, 1, NPOS, CINT, 2);
    gemm(kerB, CINT, HB, CINT, KER, nullptr, NFREQ, s_ker_b, nullptr, NFREQ, NPOS, CINT, 2);

    // ---- harmonic source ----
    fs_block_sums<<<NPOS, 256, 0, stream>>>(f0, PART);
    scan_partials<<<BB, 32, 0, stream>>>(PART, PHS);
    harm_fill<<<NPOS, 256, 0, stream>>>(f0, AMPS, PHS, DSP);

    // ---- filtered noise ----
    twiddle_init<<<(NFFT + 255) / 256, 256, 0, stream>>>(TW);
    gen_xp<<<(BB * LPAD + 255) / 256, 256, 0, stream>>>(XP);
    stft_kernel<<<dim3(LFf + 1, BB), 256, 0, stream>>>(XP, TW, KER, SPEC);
    istft_kernel<<<dim3(LFf + 1, BB), 256, 0, stream>>>(SPEC, TW, FT);
    ola_add<<<(BB * LPAD + 255) / 256, 256, 0, stream>>>(FT, TW, DSP);

    // ---- filter net trunk ----
    gemm(fInWb, CINP, XINb, CINP, XS, nullptr, CINT, f_in_b, nullptr, CINT, NPOS, CINP, 0);
    for (int i = 0; i < 6; ++i) {
        dwconv_k<<<(NPOS * CINT + 255) / 256, 256, 0, stream>>>(
            XS, f_dw_w + (size_t)i * CINT * 7, f_dw_b + (size_t)i * CINT, H);
        ln(H, nullptr, HB, f_ln_p + (size_t)i * 2 * CINT);
        gemm(fPw1b + (size_t)i * 1536 * CINT, CINT, HB, CINT, nullptr, H2b, 1536,
             f_pw1_b + (size_t)i * 1536, nullptr, 1536, NPOS, CINT, 1);
        gemm(fPw2b + (size_t)i * CINT * 1536, 1536, H2b, 1536, XS, nullptr, CINT,
             f_pw2_b + (size_t)i * CINT, XS, CINT, NPOS, 1536, 0);
    }
    ln(XS, nullptr, HB, f_out_norm);

    // ---- K=7 output conv as im2col + WMMA GEMM ----
    im2col_edge<<<(NPOS * 3584 + 255) / 256, 256, 0, stream>>>(HB, XEb);
    gemm(W2b, 3584, XEb, 3584, FILT, nullptr, NWIN, f_out_b, nullptr, NWIN, NPOS, 3584, 0);

    // ---- per-frame FIR + fold into final output ----
    fir_fold<<<NPOS, 256, 0, stream>>>(DSP, FILT, (float*)d_out);
}
